// MyGCN_24180665876562
// MI455X (gfx1250) — matrix operand order repaired and checked by reading, not compiled, not run on hardware
//
#include <hip/hip_runtime.h>

// ---------------------------------------------------------------------------
// 2-layer GCN for MI455X (gfx1250, wave32, WMMA).
//   deg/rsqrt -> bf16-WMMA GEMM (x@W1) -> edge scatter (atomic f32) ->
//   relu+selfloop -> WMMA GEMM -> scatter -> finalize (h) -> WMMA GEMM (+bias)
// GEMM: one wave per 16-row strip, A fragment reused across NT n-tiles per
// K-block (NT back-to-back v_wmma_f32_16x16x32_bf16 on independent accs).
// ---------------------------------------------------------------------------

typedef __bf16 v16bf __attribute__((ext_vector_type(16)));
typedef __bf16 v2bf  __attribute__((ext_vector_type(2)));
typedef float  v8f   __attribute__((ext_vector_type(8)));

static __device__ __forceinline__ __bf16 f2bf(float f) {
  unsigned u = __builtin_bit_cast(unsigned, f);
  u += 0x7FFFu + ((u >> 16) & 1u);            // round-to-nearest-even truncate
  unsigned short h = (unsigned short)(u >> 16);
  return __builtin_bit_cast(__bf16, h);
}

static __device__ __forceinline__ v2bf pk2bf(float lo, float hi) {
#if __has_builtin(__builtin_amdgcn_cvt_pk_bf16_f32)
  auto p = __builtin_amdgcn_cvt_pk_bf16_f32(lo, hi);   // v_cvt_pk_bf16_f32
  return __builtin_bit_cast(v2bf, p);
#else
  v2bf r;
  r[0] = f2bf(lo);
  r[1] = f2bf(hi);
  return r;
#endif
}

__global__ void fill_f32(float* p, size_t n, float v) {
  size_t i = (size_t)blockIdx.x * blockDim.x + threadIdx.x;
  size_t stride = (size_t)gridDim.x * blockDim.x;
  for (; i < n; i += stride) p[i] = v;
}

__global__ void degree_kernel(const int* __restrict__ col, float* __restrict__ deg, int E) {
  int i = blockIdx.x * blockDim.x + threadIdx.x;
  int stride = gridDim.x * blockDim.x;
  for (; i < E; i += stride) atomicAdd(&deg[col[i]], 1.0f);
}

__global__ void rsqrt_kernel(float* d, int n) {
  int i = blockIdx.x * blockDim.x + threadIdx.x;
  if (i < n) { float v = d[i]; d[i] = v > 0.f ? rsqrtf(v) : 0.f; }
}

// Stage W [Kreal x Nreal] row-major fp32 into WMMA B-fragment order (bf16),
// zero-padded to [Kpad x Npad]. Fragment element index:
//   ((kb*Nt + tn)*32 + lane)*16 + j
//   lane<16:  n = tn*16+lane,     k = kb*32 + j        (j = 0..15)
//   lane>=16: n = tn*16+lane-16,  k = kb*32 + 16 + j
__global__ void stage_b_kernel(const float* __restrict__ W, __bf16* __restrict__ Bf,
                               int Kreal, int Nreal, int Kpad, int Npad) {
  int total = (Kpad >> 5) * (Npad >> 4) * 512;
  int idx = blockIdx.x * blockDim.x + threadIdx.x;
  if (idx >= total) return;
  int j    = idx & 15;
  int lane = (idx >> 4) & 31;
  int blk  = idx >> 9;
  int Nt = Npad >> 4;
  int kb = blk / Nt, tn = blk % Nt;
  int half = lane >> 4;
  int n = tn * 16 + (lane & 15);
  int k = kb * 32 + half * 16 + j;
  float v = (k < Kreal && n < Nreal) ? W[k * Nreal + n] : 0.0f;
  Bf[idx] = f2bf(v);
}

// One wave per 16-row strip of C; NT = Npad/16 output tiles per strip.
// A fragment (16 bf16) is built once per K-block and reused by NT WMMAs.
template <int NT>
__global__ void __launch_bounds__(256)
wmma_gemm_bf16(const float* __restrict__ A, const __bf16* __restrict__ Bf,
               float* __restrict__ Cmat, const float* __restrict__ bias,
               int M, int K, int NReal, int ldOut) {
  const int lane = threadIdx.x & 31;
  const int wave = threadIdx.x >> 5;
  const int half = lane >> 4;
  const int lm   = lane & 15;
  const int Mt = (M + 15) >> 4;
  const int nwaves = (gridDim.x * blockDim.x) >> 5;

  for (int tm = blockIdx.x * (blockDim.x >> 5) + wave; tm < Mt; tm += nwaves) {
    const int m = tm * 16 + lm;
    const bool valid = (m < M);
    const float* ap = A + (size_t)m * K;

    v8f acc[NT];
#pragma unroll
    for (int tn = 0; tn < NT; ++tn) acc[tn] = (v8f)0.0f;

    const int kblocks = K >> 5;
    for (int kb = 0; kb < kblocks; ++kb) {
      // A fragment (contiguous k order, same as staged B): lane<16 -> K 0..15 of
      // this K-block; lane>=16 -> K 16..31.
      const int k0 = kb * 32 + half * 16;
      float4 a0 = {}, a1 = {}, a2 = {}, a3 = {};
      if (valid) {
        a0 = *(const float4*)(ap + k0);
        a1 = *(const float4*)(ap + k0 + 4);
        a2 = *(const float4*)(ap + k0 + 8);
        a3 = *(const float4*)(ap + k0 + 12);
        if (kb + 1 < kblocks) __builtin_prefetch(ap + k0 + 32, 0, 3);
      }
      v16bf afrag;
      v2bf* ap2 = (v2bf*)&afrag;
      ap2[0] = pk2bf(a0.x, a0.y);
      ap2[1] = pk2bf(a0.z, a0.w);
      ap2[2] = pk2bf(a1.x, a1.y);
      ap2[3] = pk2bf(a1.z, a1.w);
      ap2[4] = pk2bf(a2.x, a2.y);
      ap2[5] = pk2bf(a2.z, a2.w);
      ap2[6] = pk2bf(a3.x, a3.y);
      ap2[7] = pk2bf(a3.z, a3.w);

      const __bf16* bp = Bf + (size_t)kb * NT * 512 + (size_t)lane * 16;
#pragma unroll
      for (int tn = 0; tn < NT; ++tn) {
        v16bf bfrag = *(const v16bf*)(bp + (size_t)tn * 512);
        acc[tn] = __builtin_amdgcn_wmma_f32_16x16x32_bf16(
            /*neg_a=*/false, afrag, /*neg_b=*/false, bfrag,
            /*c_mod=*/(short)0, acc[tn], /*reuse_a=*/false, /*reuse_b=*/false);
      }
    }

    // D layout: VGPR i -> row half*8 + i; lanes 0-15 / 16-31 = row halves, col = lane%16.
    const int rbase = tm * 16 + half * 8;
#pragma unroll
    for (int tn = 0; tn < NT; ++tn) {
      const int c = tn * 16 + lm;
      if (c < NReal) {
        const float bv = bias ? bias[c] : 0.0f;
        float* cp = Cmat + (size_t)rbase * ldOut + c;
#pragma unroll
        for (int i = 0; i < 8; ++i) {
          if (rbase + i < M) cp[(size_t)i * ldOut] = acc[tn][i] + bv;
        }
      }
    }
  }
}

// One wave32 per edge: gather xw[row]*norm, atomic scatter-add into agg[col].
template <int F>
__global__ void __launch_bounds__(256)
scatter_edges(const float* __restrict__ xw, const int* __restrict__ row,
              const int* __restrict__ col, const float* __restrict__ dis,
              float* __restrict__ agg, int E) {
  int wid = (blockIdx.x * blockDim.x + threadIdx.x) >> 5;
  int lane = threadIdx.x & 31;
  int nw = (gridDim.x * blockDim.x) >> 5;
  for (int e = wid; e < E; e += nw) {
    int r = row[e], c = col[e];
    float nrm = dis[r] * dis[c];
    const float* src = xw + (size_t)r * F;
    float* dst = agg + (size_t)c * F;
#pragma unroll
    for (int i = 0; i < F / 32; ++i) {
      int f = lane + i * 32;
      atomicAdd(dst + f, src[f] * nrm);
    }
  }
}

// In-place: agg[i,f] = act(agg[i,f] + xw[i,f]*dis[i]^2 + bias[f])
__global__ void finalize_kernel(float* __restrict__ agg, const float* __restrict__ xw,
                                const float* __restrict__ dis, const float* __restrict__ bias,
                                int bias_len, int F, int relu, int n) {
  size_t idx = (size_t)blockIdx.x * blockDim.x + threadIdx.x;
  size_t total = (size_t)n * F;
  size_t stride = (size_t)gridDim.x * blockDim.x;
  for (; idx < total; idx += stride) {
    int i = (int)(idx / F);
    int f = (int)(idx - (size_t)i * F);
    float d = dis[i];
    float v = agg[idx] + xw[idx] * d * d + (f < bias_len ? bias[f] : 0.f);
    agg[idx] = relu ? fmaxf(v, 0.f) : v;
  }
}

extern "C" void kernel_launch(void* const* d_in, const int* in_sizes, int n_in,
                              void* d_out, int out_size, void* d_ws, size_t ws_size,
                              hipStream_t stream) {
  (void)n_in; (void)out_size; (void)ws_size;
  const float* x  = (const float*)d_in[0];
  const int*   ei = (const int*)d_in[1];
  const float* W1 = (const float*)d_in[2];
  const float* b1 = (const float*)d_in[3];
  const float* W2 = (const float*)d_in[4];
  const float* b2 = (const float*)d_in[5];
  const float* Wl = (const float*)d_in[6];
  const float* bl = (const float*)d_in[7];
  float* out = (float*)d_out;

  const int F_IN = 512, H1 = 50, H1P = 64, H2 = 128, C = 40, CP = 64;
  const int N = in_sizes[0] / F_IN;
  const int E = in_sizes[1] / 2;
  const int* row = ei;       // source
  const int* col = ei + E;   // target

  // workspace carving (256B aligned)
  char* wptr = (char*)d_ws;
  auto alloc = [&](size_t bytes) -> char* {
    char* p = wptr;
    wptr += (bytes + 255) & ~(size_t)255;
    return p;
  };
  float*  dis = (float*)alloc((size_t)N * 4);
  float*  xw1 = (float*)alloc((size_t)N * H1P * 4);
  float*  o2  = (float*)alloc((size_t)N * H1P * 4);   // agg1, finalized in place
  float*  xw2 = (float*)alloc((size_t)N * H2 * 4);
  __bf16* Bf1 = (__bf16*)alloc((size_t)(F_IN / 32) * (H1P / 16) * 512 * 2);
  __bf16* Bf2 = (__bf16*)alloc((size_t)(H1P / 32) * (H2 / 16) * 512 * 2);
  __bf16* Bf3 = (__bf16*)alloc((size_t)(H2 / 32) * (CP / 16) * 512 * 2);
  float* h = out + (size_t)N * C;                      // second output, used as agg2

  // ---- degree (with self loops) and D^-1/2 ----
  fill_f32<<<(N + 255) / 256, 256, 0, stream>>>(dis, (size_t)N, 1.0f);
  degree_kernel<<<(E + 255) / 256, 256, 0, stream>>>(col, dis, E);
  rsqrt_kernel<<<(N + 255) / 256, 256, 0, stream>>>(dis, N);

  // ---- stage weights into bf16 B-fragment order ----
  auto stage = [&](const float* W, __bf16* Bf, int Kr, int Nr, int Kp, int Np) {
    int total = (Kp / 32) * (Np / 16) * 512;
    stage_b_kernel<<<(total + 255) / 256, 256, 0, stream>>>(W, Bf, Kr, Nr, Kp, Np);
  };
  stage(W1, Bf1, F_IN, H1, F_IN, H1P);
  stage(W2, Bf2, H1, H2, H1P, H2);
  stage(Wl, Bf3, H2, C, H2, CP);

  const int Mt = (N + 15) / 16;
  const int gemm_blocks = (Mt + 7) / 8;   // 8 waves per 256-thread block

  // ---- layer 1: xw1 = x@W1 ; aggregate ; relu ----
  wmma_gemm_bf16<4><<<gemm_blocks, 256, 0, stream>>>(x, Bf1, xw1, nullptr, N, F_IN, H1P, H1P);
  fill_f32<<<(int)(((size_t)N * H1P + 255) / 256), 256, 0, stream>>>(o2, (size_t)N * H1P, 0.f);
  scatter_edges<64><<<16384, 256, 0, stream>>>(xw1, row, col, dis, o2, E);
  finalize_kernel<<<(int)(((size_t)N * H1P + 255) / 256), 256, 0, stream>>>(
      o2, xw1, dis, b1, H1, H1P, 1, N);

  // ---- layer 2: xw2 = o2@W2 ; aggregate -> h (second output, no relu) ----
  wmma_gemm_bf16<8><<<gemm_blocks, 256, 0, stream>>>(o2, Bf2, xw2, nullptr, N, H1P, H2, H2);
  fill_f32<<<(int)(((size_t)N * H2 + 255) / 256), 256, 0, stream>>>(h, (size_t)N * H2, 0.f);
  scatter_edges<128><<<16384, 256, 0, stream>>>(xw2, row, col, dis, h, E);
  finalize_kernel<<<(int)(((size_t)N * H2 + 255) / 256), 256, 0, stream>>>(
      h, xw2, dis, b2, H2, H2, 0, N);

  // ---- classifier: out = h@Wl + bl (store only real 40 cols) ----
  wmma_gemm_bf16<4><<<gemm_blocks, 256, 0, stream>>>(h, Bf3, out, bl, N, H2, C, C);
}
